// AttentionPool_82171314307530
// MI455X (gfx1250) — matrix operand order repaired and checked by reading, not compiled, not run on hardware
//
#include <hip/hip_runtime.h>
#include <math.h>

// ---------------------------------------------------------------------------
// AttentionPool for MI455X (gfx1250, wave32, WMMA).
//   scores = tanh(h @ W^T + b) @ context          (bf16-split WMMA, ~fp32 acc)
//   per-segment softmax (blk_ptr boundaries)  ->  alpha
//   h_file[k] = sum_{i in seg k} alpha_i * h_i
// Outputs: d_out = [h_file (B*128 f32)] ++ [alpha (N f32)]
// ---------------------------------------------------------------------------

typedef __attribute__((ext_vector_type(16))) __bf16 v16bf;
typedef __attribute__((ext_vector_type(8)))  float  v8f;

#define DIM 128

__device__ __forceinline__ unsigned short f2bfu(float f) {
  union { float f; unsigned u; } v; v.f = f;
  unsigned r = v.u + 0x7FFFu + ((v.u >> 16) & 1u);   // round-to-nearest-even
  return (unsigned short)(r >> 16);
}
__device__ __forceinline__ float bfu2f(unsigned short hs) {
  union { unsigned u; float f; } v; v.u = ((unsigned)hs) << 16; return v.f;
}
__device__ __forceinline__ unsigned f2u(float f) {
  union { float f; unsigned u; } v; v.f = f; return v.u;
}
__device__ __forceinline__ float u2f(unsigned u) {
  union { unsigned u; float f; } v; v.u = u; return v.f;
}

// Branch-free tanh: no EXEC divergence near WMMA, 2 transcendentals.
// tanh(x) = sign(x) * (1 - e) / (1 + e),  e = exp(-2|x|)
__device__ __forceinline__ float fast_tanh(float x) {
  float ax = fabsf(x);
  float e  = __expf(-2.0f * ax);
  float r  = __fdividef(1.0f - e, 1.0f + e);
  return copysignf(r, x);
}

// ---------------- Kernel 0: split W (fp32) into bf16 hi/lo ----------------
__global__ void wsplit_kernel(const float* __restrict__ W,
                              unsigned short* __restrict__ Wh,
                              unsigned short* __restrict__ Wl, int n) {
  int i = blockIdx.x * blockDim.x + threadIdx.x;
  if (i < n) {
    float w = W[i];
    unsigned short hh = f2bfu(w);
    Wh[i] = hh;
    Wl[i] = f2bfu(w - bfu2f(hh));
  }
}

// ---------------- Kernel 1: scores via bf16-split WMMA --------------------
// One wave per 16-row strip of h. 256 threads = 8 waves/block.
// A (16x32 bf16) lane layout: lane&15 -> M, K = kb..kb+15 (contiguous)
// with kb = kg*32 (+16 for lanes>=16); same K order as B.
// B (32x16 bf16): lane&15 = column N, lanes<16 hold K=0..15, lanes>=16 K=16..31,
// K-contiguous -> 16 contiguous bf16 per lane from row n of W.
__global__ __launch_bounds__(256) void scores_kernel(
    const float* __restrict__ h, const unsigned short* __restrict__ Wh,
    const unsigned short* __restrict__ Wl, const float* __restrict__ bias,
    const float* __restrict__ ctx, float* __restrict__ scores, int nstrips) {
  const int lane  = threadIdx.x & 31;
  const int wave  = threadIdx.x >> 5;
  const int strip = blockIdx.x * 8 + wave;
  if (strip >= nstrips) return;                 // wave-uniform: EXEC stays full

  const int r0  = strip * 16;
  const int lm  = lane & 15;
  const int hiL = lane >> 4;                    // 0 or 1
  const int row = r0 + lm;

  // Build A tiles (hi/lo bf16) for all four K-groups of 32.
  // hi = truncate-to-bf16(x) (mask), lo = truncate-to-bf16(x - hi); pairs of
  // high halves packed with one v_perm_b32 each (~3 VALU/element).
  union { v16bf bf; unsigned u[8]; } AhU[4], AlU[4];
  const float* hrow = h + (size_t)row * DIM;
#pragma unroll
  for (int kg = 0; kg < 4; ++kg) {
    const int kb = kg * 32 + hiL * 16;
#pragma unroll
    for (int half = 0; half < 2; ++half) {      // elems 0..7 -> K=kb.., 8..15 -> K=kb+8..
      const float* p = hrow + kb + half * 8;
#pragma unroll
      for (int e = 0; e < 8; e += 2) {
        float x0 = p[e], x1 = p[e + 1];
        unsigned u0 = f2u(x0), u1 = f2u(x1);
        float l0 = x0 - u2f(u0 & 0xFFFF0000u);  // exact
        float l1 = x1 - u2f(u1 & 0xFFFF0000u);  // exact
        AhU[kg].u[half * 4 + (e >> 1)] =
            __builtin_amdgcn_perm(u1, u0, 0x07060302u);
        AlU[kg].u[half * 4 + (e >> 1)] =
            __builtin_amdgcn_perm(f2u(l1), f2u(l0), 0x07060302u);
      }
    }
  }

  float part[8];
#pragma unroll
  for (int v = 0; v < 8; ++v) part[v] = 0.f;

#pragma unroll 1
  for (int nt = 0; nt < 8; ++nt) {              // 8 tiles of 16 output features
    const int n = nt * 16 + lm;
    v8f acc = {};
#pragma unroll
    for (int kg = 0; kg < 4; ++kg) {
      const int koff = kg * 32 + hiL * 16;
      v16bf Bh = *(const v16bf*)(Wh + (size_t)n * DIM + koff);
      v16bf Bl = *(const v16bf*)(Wl + (size_t)n * DIM + koff);
      // x*w ~= ah*bh + ah*bl + al*bh  (drop al*bl ~ 2^-16)
      acc = __builtin_amdgcn_wmma_f32_16x16x32_bf16(false, AhU[kg].bf, false, Bh,
                                                    (short)0, acc, false, false);
      acc = __builtin_amdgcn_wmma_f32_16x16x32_bf16(false, AhU[kg].bf, false, Bl,
                                                    (short)0, acc, false, false);
      acc = __builtin_amdgcn_wmma_f32_16x16x32_bf16(false, AlU[kg].bf, false, Bh,
                                                    (short)0, acc, false, false);
    }
    // D tile: VGPR v -> (M = v + 8*hiL, N = n). Fold tanh + context dot.
    const float nb = bias[n];
    const float nc = ctx[n];
#pragma unroll
    for (int v = 0; v < 8; ++v) part[v] += nc * fast_tanh(acc[v] + nb);
  }

  // Reduce the per-column partials across each 16-lane half (wave32 shuffles).
#pragma unroll
  for (int m = 1; m <= 8; m <<= 1) {
#pragma unroll
    for (int v = 0; v < 8; ++v) part[v] += __shfl_xor(part[v], m, 32);
  }
  if (lm == 0) {
#pragma unroll
    for (int v = 0; v < 8; ++v) scores[r0 + hiL * 8 + v] = part[v];
  }
}

// ---------------- Kernel 1b: scalar tail (rows not covered by strips) -----
__global__ void scores_tail_kernel(const float* __restrict__ h,
                                   const float* __restrict__ W,
                                   const float* __restrict__ bias,
                                   const float* __restrict__ ctx,
                                   float* __restrict__ scores, int base) {
  __shared__ float red[DIM];
  const int i = base + blockIdx.x;              // row
  const int j = threadIdx.x;                    // feature
  float t = 0.f;
  for (int k = 0; k < DIM; ++k) t += h[(size_t)i * DIM + k] * W[(size_t)j * DIM + k];
  red[j] = ctx[j] * fast_tanh(t + bias[j]);
  __syncthreads();
  for (int off = DIM / 2; off > 0; off >>= 1) {
    if (j < off) red[j] += red[j + off];
    __syncthreads();
  }
  if (j == 0) scores[i] = red[0];
}

// ---------------- Kernel 2: segment softmax + weighted pool ---------------
// One 128-thread block per segment. `alpha` buffer initially holds scores
// (written by kernel 1) and is overwritten in place with alpha.
__global__ __launch_bounds__(128) void pool_kernel(
    const float* __restrict__ h, const int* __restrict__ ptr,
    float* __restrict__ h_file, float* __restrict__ alpha) {
  __shared__ float red[DIM];
  __shared__ float ebuf[DIM];
  const int tid = threadIdx.x;
  const int k = blockIdx.x;
  const int s = ptr[k];
  const int e = ptr[k + 1];

  // pass 1: segment max
  float mx = -INFINITY;
  for (int i = s + tid; i < e; i += DIM) mx = fmaxf(mx, alpha[i]);
  red[tid] = mx; __syncthreads();
  for (int off = DIM / 2; off > 0; off >>= 1) {
    if (tid < off) red[tid] = fmaxf(red[tid], red[tid + off]);
    __syncthreads();
  }
  mx = red[0]; __syncthreads();

  // pass 2: sum of exp
  float sum = 0.f;
  for (int i = s + tid; i < e; i += DIM) sum += __expf(alpha[i] - mx);
  red[tid] = sum; __syncthreads();
  for (int off = DIM / 2; off > 0; off >>= 1) {
    if (tid < off) red[tid] += red[tid + off];
    __syncthreads();
  }
  const float denom = red[0];
  const float inv = denom > 0.f ? 1.f / denom : 0.f;
  __syncthreads();

  // pass 3: alpha (in place) + weighted row-sum, thread-per-dim
  float acc = 0.f;
  for (int c = s; c < e; c += DIM) {
    const int i = c + tid;
    float ev = 0.f;
    if (i < e) {
      ev = __expf(alpha[i] - mx) * inv;
      alpha[i] = ev;                            // own element: no race
    }
    ebuf[tid] = ev;
    __syncthreads();
    const int cnt = min(DIM, e - c);
    for (int j = 0; j < cnt; ++j)
      acc += ebuf[j] * h[(size_t)(c + j) * DIM + tid];
    __syncthreads();
  }
  h_file[(size_t)k * DIM + tid] = acc;          // empty segment -> zeros
}

// ---------------------------------------------------------------------------
extern "C" void kernel_launch(void* const* d_in, const int* in_sizes, int n_in,
                              void* d_out, int out_size, void* d_ws, size_t ws_size,
                              hipStream_t stream) {
  const float* h   = (const float*)d_in[0];   // [N,128]
  const float* W   = (const float*)d_in[1];   // [128,128]
  const float* b   = (const float*)d_in[2];   // [128]
  const float* ctx = (const float*)d_in[3];   // [128]
  const int*   ptr = (const int*)d_in[4];     // [B+1]

  const int N = in_sizes[0] / DIM;
  const int B = in_sizes[4] - 1;

  unsigned short* Wh = (unsigned short*)d_ws;           // 32 KB
  unsigned short* Wl = Wh + DIM * DIM;                  // 32 KB

  float* out    = (float*)d_out;
  float* h_file = out;                                  // [B,128]
  float* alpha  = out + (size_t)B * DIM;                // [N]; holds scores first

  // 0) split W into bf16 hi/lo
  wsplit_kernel<<<(DIM * DIM + 255) / 256, 256, 0, stream>>>(W, Wh, Wl, DIM * DIM);

  // 1) scores via WMMA (16-row strips per wave)
  const int nstrips = N / 16;
  if (nstrips > 0)
    scores_kernel<<<(nstrips + 7) / 8, 256, 0, stream>>>(h, Wh, Wl, b, ctx,
                                                         alpha, nstrips);
  const int rem = N - nstrips * 16;
  if (rem > 0)
    scores_tail_kernel<<<rem, DIM, 0, stream>>>(h, W, b, ctx, alpha, nstrips * 16);

  // 2) per-segment softmax + pooled rows
  pool_kernel<<<B, DIM, 0, stream>>>(h, ptr, h_file, alpha);
}
